// GemmaAttention_81028853006549
// MI455X (gfx1250) — compile-verified, matrix operand order checked by reading, not run on hardware
//
#include <hip/hip_runtime.h>
#include <hip/hip_bf16.h>
#include <cstdint>

// ---------------------------------------------------------------------------
// GemmaAttention forward for MI455X (gfx1250, wave32, WMMA).
// - all matmuls: v_wmma_f32_16x16x32_bf16 (bf16 in, f32 accumulate)
// - fragments loaded as 2x b128 per lane (B operands pre-transposed)
// - fused scores+softmax+PV kernel: attn_weights written to HBM exactly once,
//   P kept in LDS (bf16) for the PV phase; Q strip staged in LDS via
//   global_load_async_to_lds_b128 / s_wait_asynccnt (CDNA5 async path)
// ---------------------------------------------------------------------------

typedef __attribute__((ext_vector_type(16))) __bf16         v16bf;
typedef __attribute__((ext_vector_type(16))) unsigned short v16us;
typedef __attribute__((ext_vector_type(8)))  float          v8f;
typedef unsigned short bfbits;

struct frag32 { uint4 lo; uint4 hi; };   // 32 bytes == v16bf

#define B_  2
#define S_  2048
#define H_  2048
#define NH_ 8
#define D_  256
#define D2_ 128
#define BS_ (B_ * S_)
#define SCALING_ 0.0625f      // 256^-0.5
#define NEG_INF_ (-1000000000.0f)

#define QS_LD 264             // qslab row stride (elems): 4*m banks, 16B aligned
#define PS_LD 2056            // pslab row stride (elems)

// ---- float -> bf16 (round to nearest even) as raw bits ---------------------
__device__ __forceinline__ unsigned short f2bf_bits(float f) {
  unsigned u = __float_as_uint(f);
  u += 0x7fffu + ((u >> 16) & 1u);
  return (unsigned short)(u >> 16);
}

// ---- A-style fragment (16x32 bf16 tile, row-major src, global or LDS) ------
// ISA 7.12.2 (16-bit A): lane 0-15 -> M, +8 element offset for lanes 16-31;
// per lane: elements {0..7} and {16..23} of the k0 window -> two 16B loads.
__device__ __forceinline__ v16bf load_frag_a(const bfbits* A, int lda,
                                             int row0, int k0, int lane) {
  int m    = row0 + (lane & 15);
  int half = (lane >> 4) & 1;
  const bfbits* p = A + (size_t)m * lda + k0 + half * 8;
  frag32 f;
  f.lo = *(const uint4*)(p);
  f.hi = *(const uint4*)(p + 16);
  return __builtin_bit_cast(v16bf, f);
}

// ---- async copy of one 16B chunk into LDS (CDNA5 Tensor/async path) --------
typedef __attribute__((address_space(3))) const void* lds_cptr_t;
__device__ __forceinline__ void async_copy_b128(const void* gsrc, void* ldst) {
  unsigned off = (unsigned)(uintptr_t)(lds_cptr_t)ldst;
  asm volatile("global_load_async_to_lds_b128 %0, %1, off"
               :: "v"(off), "v"(gsrc) : "memory");
}
__device__ __forceinline__ void wait_async0() {
  asm volatile("s_wait_asynccnt 0" ::: "memory");
}

// ---------------------------------------------------------------------------
// f32 -> bf16 bits elementwise (A operands, layouts preserved)
// ---------------------------------------------------------------------------
__global__ void cvt_f32_bf16(const float* __restrict__ src,
                             bfbits* __restrict__ dst, int n) {
  int i = blockIdx.x * blockDim.x + threadIdx.x;
  if (i < n) dst[i] = f2bf_bits(src[i]);
}

// ---------------------------------------------------------------------------
// Transpose + convert: src (R x C, f32) -> dst (C x R, bf16). 32x32 LDS tile.
// ---------------------------------------------------------------------------
__global__ __launch_bounds__(256) void transpose_cvt(const float* __restrict__ src,
                                                     bfbits* __restrict__ dst,
                                                     int R, int C) {
  __shared__ bfbits tile[32][33];
  int tx = threadIdx.x & 31;
  int ty = threadIdx.x >> 5;
  int r0 = blockIdx.y * 32;
  int c0 = blockIdx.x * 32;
#pragma unroll
  for (int i = 0; i < 32; i += 8)
    tile[ty + i][tx] = f2bf_bits(src[(size_t)(r0 + ty + i) * C + c0 + tx]);
  __syncthreads();
#pragma unroll
  for (int i = 0; i < 32; i += 8)
    dst[(size_t)(c0 + ty + i) * R + r0 + tx] = tile[tx][ty + i];
}

// ---------------------------------------------------------------------------
// GEMM: C(MxN, f32) = A(MxK, bf16) @ BT(NxK, bf16)^T
// Wave computes 16x128 strip. Fragment loads batched ahead of the WMMA burst
// so loads cluster and overlap compute (operands are L2-resident).
// ---------------------------------------------------------------------------
__global__ __launch_bounds__(256) void gemm_bf16(const bfbits* __restrict__ A,
                                                 const bfbits* __restrict__ BT,
                                                 float* __restrict__ C,
                                                 int M, int N, int K) {
  int wave = blockIdx.x * (blockDim.x >> 5) + (threadIdx.x >> 5);
  int lane = threadIdx.x & 31;
  int ng   = N >> 7;
  int mt   = wave / ng;
  int cg   = wave - mt * ng;
  if (mt * 16 >= M) return;
  int row0 = mt * 16, col0 = cg * 128;

  v8f acc[8] = {};
#pragma unroll 2
  for (int k0 = 0; k0 < K; k0 += 32) {
    v16bf a = load_frag_a(A, K, row0, k0, lane);
    __builtin_prefetch(A + (size_t)(row0 + (lane & 15)) * K + k0 + 64, 0, 1);
    v16bf bf[8];
#pragma unroll
    for (int j = 0; j < 8; ++j)
      bf[j] = load_frag_a(BT, K, col0 + 16 * j, k0, lane);
#pragma unroll
    for (int j = 0; j < 8; ++j)
      acc[j] = __builtin_amdgcn_wmma_f32_16x16x32_bf16(
          false, a, false, bf[j], (short)0, acc[j], false, false);
  }
  int half = (lane >> 4) & 1, n = lane & 15;
#pragma unroll
  for (int j = 0; j < 8; ++j)
#pragma unroll
    for (int r = 0; r < 8; ++r)
      C[(size_t)(row0 + r + half * 8) * N + col0 + 16 * j + n] = acc[j][r];
}

// ---------------------------------------------------------------------------
// RoPE epilogues
// ---------------------------------------------------------------------------
__global__ void rope_q(const float* __restrict__ Qf,
                       const float* __restrict__ cosp,
                       const float* __restrict__ sinp,
                       bfbits* __restrict__ Qb) {
  int i = blockIdx.x * blockDim.x + threadIdx.x;  // 2^23
  int d = i & (D_ - 1);
  int h = (i >> 8) & (NH_ - 1);
  int s = (i >> 11) & (S_ - 1);
  int b = i >> 22;
  const float* qrow = Qf + (size_t)(b * S_ + s) * (NH_ * D_) + h * D_;
  float x     = qrow[d];
  float other = qrow[d < D2_ ? d + D2_ : d - D2_];
  float rot   = (d < D2_) ? -other : other;
  size_t cs   = (size_t)(b * S_ + s) * D_ + d;
  float val   = x * cosp[cs] + rot * sinp[cs];
  Qb[(((size_t)(b * NH_ + h) * S_) + s) * D_ + d] = f2bf_bits(val);
}

__global__ void rope_k(const float* __restrict__ Kf,
                       const float* __restrict__ cosp,
                       const float* __restrict__ sinp,
                       bfbits* __restrict__ Kb) {
  int i = blockIdx.x * blockDim.x + threadIdx.x;  // 2^20
  int d = i & (D_ - 1);
  int s = (i >> 8) & (S_ - 1);
  int b = i >> 19;
  const float* krow = Kf + (size_t)(b * S_ + s) * D_;
  float x     = krow[d];
  float other = krow[d < D2_ ? d + D2_ : d - D2_];
  float rot   = (d < D2_) ? -other : other;
  size_t cs   = (size_t)(b * S_ + s) * D_ + d;
  float val   = x * cosp[cs] + rot * sinp[cs];
  Kb[(size_t)(b * S_ + s) * D_ + d] = f2bf_bits(val);
}

// ---------------------------------------------------------------------------
// Fused attention core. Block = 8 waves = one 16-row q strip.
//  phase 0: async-stage the Q strip (16x256 bf16) into LDS
//  phase 1: each wave computes 16x256 logits (16 accs) over all 2048 keys
//  softmax: block-wide row max/sum via LDS reduction array
//  write:   normalized P -> d_out (f32, the attn_weights output, written once)
//           and bf16 copy into LDS slab
//  phase 2: attn = P @ V from the LDS slab (ds_load_b128 A-fragments)
// ---------------------------------------------------------------------------
__global__ __launch_bounds__(256) void attn_fused(const bfbits* __restrict__ Qb,
                                                  const bfbits* __restrict__ Kb,
                                                  const bfbits* __restrict__ VTb,
                                                  float* __restrict__ Pout,
                                                  bfbits* __restrict__ Ob) {
  __shared__ bfbits qslab[16 * QS_LD];   //  8.4 KB
  __shared__ bfbits pslab[16 * PS_LD];   // 65.8 KB
  __shared__ float  red[16][129];        //  8.3 KB

  int t    = threadIdx.x;
  int w    = t >> 5;
  int lane = t & 31;
  int half = (lane >> 4) & 1;
  int n    = lane & 15;
  int qt   = blockIdx.x;          // 128 q tiles
  int bh   = blockIdx.y;          // 16 (b,h)
  int b    = bh >> 3, h = bh & 7;
  int row0g = qt * 16;

  // ---- phase 0: async copy Q strip (16 rows x 256 d) to LDS ----------------
  const bfbits* Qp = Qb + ((size_t)bh * S_ + row0g) * D_;
#pragma unroll
  for (int c = 0; c < 2; ++c) {
    int chunk = t + 256 * c;              // 512 chunks of 16B
    int row   = chunk >> 5;
    int ch    = chunk & 31;
    async_copy_b128(Qp + (size_t)row * D_ + ch * 8,
                    &qslab[row * QS_LD + ch * 8]);
  }
  wait_async0();
  __syncthreads();

  // ---- phase 1: logits, 16 accs per wave (cols w*256 .. w*256+255) ---------
  const bfbits* Kp = Kb + (size_t)b * S_ * D_;
  int col0 = w * 256;
  v8f acc[16] = {};
#pragma unroll
  for (int k0 = 0; k0 < D_; k0 += 32) {
    v16bf a = load_frag_a(qslab, QS_LD, 0, k0, lane);
#pragma unroll
    for (int jb = 0; jb < 4; ++jb) {
      v16bf bf[4];
#pragma unroll
      for (int u = 0; u < 4; ++u)
        bf[u] = load_frag_a(Kp, D_, col0 + 16 * (jb * 4 + u), k0, lane);
#pragma unroll
      for (int u = 0; u < 4; ++u)
        acc[jb * 4 + u] = __builtin_amdgcn_wmma_f32_16x16x32_bf16(
            false, a, false, bf[u], (short)0, acc[jb * 4 + u], false, false);
    }
  }

  // ---- scale + causal mask (in registers) ----------------------------------
#pragma unroll
  for (int j = 0; j < 16; ++j)
#pragma unroll
    for (int r = 0; r < 8; ++r) {
      int qrow = row0g + r + half * 8;
      int col  = col0 + 16 * j + n;
      acc[j][r] = acc[j][r] * SCALING_ + (col > qrow ? NEG_INF_ : 0.0f);
    }

  // ---- block-wide row max ---------------------------------------------------
  float lmax[8];
#pragma unroll
  for (int r = 0; r < 8; ++r) {
    float m = acc[0][r];
#pragma unroll
    for (int j = 1; j < 16; ++j) m = fmaxf(m, acc[j][r]);
    lmax[r] = m;
  }
#pragma unroll
  for (int r = 0; r < 8; ++r) red[r + half * 8][w * 16 + n] = lmax[r];
  __syncthreads();
  if (t < 16) {
    float m = red[t][0];
    for (int c = 1; c < 128; ++c) m = fmaxf(m, red[t][c]);
    red[t][128] = m;
  }
  __syncthreads();

  // ---- exp + block-wide row sum --------------------------------------------
  float rmax[8], lsum[8];
#pragma unroll
  for (int r = 0; r < 8; ++r) { rmax[r] = red[r + half * 8][128]; lsum[r] = 0.0f; }
#pragma unroll
  for (int j = 0; j < 16; ++j)
#pragma unroll
    for (int r = 0; r < 8; ++r) {
      float e = __expf(acc[j][r] - rmax[r]);
      acc[j][r] = e;
      lsum[r] += e;
    }
  __syncthreads();
#pragma unroll
  for (int r = 0; r < 8; ++r) red[r + half * 8][w * 16 + n] = lsum[r];
  __syncthreads();
  if (t < 16) {
    float s = 0.0f;
    for (int c = 0; c < 128; ++c) s += red[t][c];
    red[t][128] = 1.0f / s;
  }
  __syncthreads();
  float rinv[8];
#pragma unroll
  for (int r = 0; r < 8; ++r) rinv[r] = red[r + half * 8][128];

  // ---- write normalized P: f32 -> d_out (once), bf16 -> LDS slab -----------
  float* Pg = Pout + (size_t)bh * S_ * S_;
#pragma unroll
  for (int j = 0; j < 16; ++j)
#pragma unroll
    for (int r = 0; r < 8; ++r) {
      int row = r + half * 8;
      int col = col0 + 16 * j + n;
      float p = acc[j][r] * rinv[r];
      Pg[(size_t)(row0g + row) * S_ + col] = p;
      pslab[row * PS_LD + col] = f2bf_bits(p);
    }
  __syncthreads();

  // ---- phase 2: attn = P @ V from LDS (wave covers d = w*32 .. w*32+31) ----
  const bfbits* Vp = VTb + (size_t)b * D_ * S_;   // (D x S)
  v8f acc2[2] = {};
#pragma unroll 2
  for (int k0 = 0; k0 < S_; k0 += 32) {
    v16bf a  = load_frag_a(pslab, PS_LD, 0, k0, lane);
    v16bf b0 = load_frag_a(Vp, S_, w * 32, k0, lane);
    v16bf b1 = load_frag_a(Vp, S_, w * 32 + 16, k0, lane);
    acc2[0] = __builtin_amdgcn_wmma_f32_16x16x32_bf16(
        false, a, false, b0, (short)0, acc2[0], false, false);
    acc2[1] = __builtin_amdgcn_wmma_f32_16x16x32_bf16(
        false, a, false, b1, (short)0, acc2[1], false, false);
  }
#pragma unroll
  for (int j = 0; j < 2; ++j)
#pragma unroll
    for (int r = 0; r < 8; ++r) {
      int row = r + half * 8;
      int d   = w * 32 + 16 * j + n;
      Ob[(size_t)(b * S_ + row0g + row) * (NH_ * D_) + h * D_ + d] =
          f2bf_bits(acc2[j][r]);
    }
}

// ---------------------------------------------------------------------------
// Host-side orchestration
// ---------------------------------------------------------------------------
extern "C" void kernel_launch(void* const* d_in, const int* in_sizes, int n_in,
                              void* d_out, int out_size, void* d_ws,
                              size_t ws_size, hipStream_t stream) {
  const float* X    = (const float*)d_in[0];
  const float* cosp = (const float*)d_in[1];
  const float* sinp = (const float*)d_in[2];
  // d_in[3] = attention_mask (causal, recomputed on the fly)
  const float* Wq = (const float*)d_in[4];
  const float* Wk = (const float*)d_in[5];
  const float* Wv = (const float*)d_in[6];
  const float* Wo = (const float*)d_in[7];

  float* out_attn = (float*)d_out;
  float* out_w    = (float*)d_out + (size_t)B_ * S_ * H_;

  bfbits* wsb = (bfbits*)d_ws;
  size_t o = 0;
  bfbits* Xb  = wsb + o; o += (size_t)BS_ * H_;
  bfbits* Wqt = wsb + o; o += (size_t)NH_ * D_ * H_;
  bfbits* Wkt = wsb + o; o += (size_t)D_ * H_;
  bfbits* Wvt = wsb + o; o += (size_t)D_ * H_;
  bfbits* Wot = wsb + o; o += (size_t)H_ * NH_ * D_;
  bfbits* Qb  = wsb + o; o += (size_t)B_ * NH_ * S_ * D_;
  bfbits* Kb  = wsb + o; o += (size_t)B_ * S_ * D_;
  bfbits* VTb = wsb + o; o += (size_t)B_ * D_ * S_;
  bfbits* Ab  = wsb + o; o += (size_t)BS_ * NH_ * D_;
  float*  Qf  = (float*)(wsb + o);
  float*  KVf = Qf + (size_t)BS_ * NH_ * D_;

  const int T = 256;

  cvt_f32_bf16<<<(BS_ * H_ + T - 1) / T, T, 0, stream>>>(X, Xb, BS_ * H_);
  transpose_cvt<<<dim3(NH_ * D_ / 32, H_ / 32), T, 0, stream>>>(Wq, Wqt, H_, NH_ * D_);
  transpose_cvt<<<dim3(D_ / 32, H_ / 32), T, 0, stream>>>(Wk, Wkt, H_, D_);
  transpose_cvt<<<dim3(D_ / 32, H_ / 32), T, 0, stream>>>(Wv, Wvt, H_, D_);
  transpose_cvt<<<dim3(H_ / 32, NH_ * D_ / 32), T, 0, stream>>>(Wo, Wot, NH_ * D_, H_);

  {
    int waves = (BS_ / 16) * ((NH_ * D_) / 128);
    gemm_bf16<<<waves / 8, T, 0, stream>>>(Xb, Wqt, Qf, BS_, NH_ * D_, H_);
    rope_q<<<(B_ * NH_ * S_ * D_) / T, T, 0, stream>>>(Qf, cosp, sinp, Qb);
  }
  {
    int waves = (BS_ / 16) * (D_ / 128);
    gemm_bf16<<<waves / 8, T, 0, stream>>>(Xb, Wkt, KVf, BS_, D_, H_);
    rope_k<<<(B_ * S_ * D_) / T, T, 0, stream>>>(KVf, cosp, sinp, Kb);
    gemm_bf16<<<waves / 8, T, 0, stream>>>(Xb, Wvt, KVf, BS_, D_, H_);
    for (int b = 0; b < B_; ++b)
      transpose_cvt<<<dim3(D_ / 32, S_ / 32), T, 0, stream>>>(
          KVf + (size_t)b * S_ * D_, VTb + (size_t)b * D_ * S_, S_, D_);
  }

  // fused scores + softmax + PV (attn_weights written exactly once)
  attn_fused<<<dim3(S_ / 16, B_ * NH_), T, 0, stream>>>(Qb, Kb, VTb, out_w, Ab);

  {
    int waves = (BS_ / 16) * (H_ / 128);
    gemm_bf16<<<waves / 8, T, 0, stream>>>(Ab, Wot, out_attn, BS_, H_, NH_ * D_);
  }

  (void)in_sizes; (void)n_in; (void)out_size; (void)ws_size;
}
